// EfficientAutoCorrelation_49160195670632
// MI455X (gfx1250) — matrix operand order repaired and checked by reading, not compiled, not run on hardware
//
#include <hip/hip_runtime.h>

typedef _Float16 h4   __attribute__((ext_vector_type(4)));
typedef _Float16 h8   __attribute__((ext_vector_type(8)));
typedef _Float16 v8h  __attribute__((ext_vector_type(8)));
typedef _Float16 v16h __attribute__((ext_vector_type(16)));
typedef float    v8f  __attribute__((ext_vector_type(8)));
typedef unsigned u32x4 __attribute__((ext_vector_type(4)));
typedef int      i32x4 __attribute__((ext_vector_type(4)));
typedef int      i32x8 __attribute__((ext_vector_type(8)));

#define B_    8
#define L_    4096
#define H_    16
#define E_    64
#define HE    1024          // H*E
#define LC    512           // chunk length
#define NC    8             // number of chunks
#define CD    8192          // reduction dim = B*H*E
#define TOPK  6
#define TILE  64            // block tile (M and N)
#define KC    64            // K step per LDS stage
#define NSTEP (CD / KC)     // 128 K-steps
#define LDST  72            // padded halfs per LDS row (144B, 16B aligned)

#define QH_BYTES ((size_t)B_ * L_ * HE * sizeof(_Float16))   // 64 MiB per tensor

// TDM availability: device pass only, and only if the builtin exists.
#if defined(__HIP_DEVICE_COMPILE__)
  #if __has_builtin(__builtin_amdgcn_tensor_load_to_lds)
    #define TDM_OK 1
  #else
    #define TDM_OK 0
  #endif
#else
  #define TDM_OK 0   // host parse: use plain-C staging branch
#endif

// ---------------------------------------------------------------------------
__global__ void zero_corr_kernel(float* __restrict__ corr) {
    int i = blockIdx.x * blockDim.x + threadIdx.x;
    if (i < NC * LC) corr[i] = 0.0f;
}

// ---------------------------------------------------------------------------
// Streaming f32 -> f16 conversion (q,k pre-pass). 8 floats per thread.
// ---------------------------------------------------------------------------
__global__ __launch_bounds__(256) void cvt_f16_kernel(
        const float* __restrict__ src, _Float16* __restrict__ dst, int n8) {
    int i = blockIdx.x * 256 + threadIdx.x;
    if (i < n8) {
        const float4 a = ((const float4*)src)[2 * i];
        const float4 b = ((const float4*)src)[2 * i + 1];
        h8 o = { (_Float16)a.x, (_Float16)a.y, (_Float16)a.z, (_Float16)a.w,
                 (_Float16)b.x, (_Float16)b.y, (_Float16)b.z, (_Float16)b.w };
        *(h8*)(dst + (size_t)i * 8) = o;
    }
}

#if TDM_OK
// Issue one 2D TDM tile load: 64 rows x 64 halfs, row stride 1024 halfs,
// LDS padded 4 DWORDs every 32 DWORDs  ->  LDS row stride = 72 halfs = LDST.
__device__ __forceinline__ void tdm_load_tile_f16(unsigned long long gaddr,
                                                  unsigned lds_addr) {
    u32x4 g0 = { 1u,                                   // count=1 (valid user D#)
                 lds_addr,                             // lds_addr
                 (unsigned)(gaddr & 0xFFFFFFFFull),    // global_addr[31:0]
                 ((unsigned)((gaddr >> 32) & 0x01FFFFFFull)) | 0x80000000u }; // type=2
    i32x8 g1 = { (int)0x07110000u,     // data_size=2B, pad_en, interval=32DW, amount=4DW
                 (int)(1024u << 16),   // tensor_dim0 = 1024 (elements)
                 (int)(4096u << 16),   // tensor_dim1 = 4096 (rows)
                 (int)(64u << 16),     // tile_dim0 = 64
                 (int)64,              // tile_dim1 = 64, tile_dim2 = 0
                 (int)1024,            // tensor_dim0_stride = 1024 (elements)
                 0, 0 };               // dim1 stride unused (2D)
    i32x4 z4 = { 0, 0, 0, 0 };
#if __clang_major__ >= 23
    i32x8 z8 = { 0, 0, 0, 0, 0, 0, 0, 0 };
    __builtin_amdgcn_tensor_load_to_lds(g0, g1, z4, z4, z8, 0);
#else
    __builtin_amdgcn_tensor_load_to_lds(g0, g1, z4, z4, 0);
#endif
}
#endif

// ---------------------------------------------------------------------------
// WMMA GEMM on pre-converted f16 tensors. Tiles are staged by the Tensor Data
// Mover with double-buffered LDS so DMA for step i+1 overlaps WMMA of step i
// (in-order TDM completion => s_wait_tensorcnt<=2 means the current buffer's
// two tiles have landed while the next pair is still in flight).
//   grid = (NC, 512/TILE, 512/TILE), block = 128 (4 wave32s).
// ---------------------------------------------------------------------------
__global__ __launch_bounds__(128) void corr_gemm_tdm_kernel(
        const _Float16* __restrict__ qh, const _Float16* __restrict__ kh,
        float* __restrict__ corr) {
    __shared__ __attribute__((aligned(16))) _Float16 Alds[2][TILE * LDST];
    __shared__ __attribute__((aligned(16))) _Float16 Blds[2][TILE * LDST];
    __shared__ float cbuf[LC];

    const int chunk = blockIdx.x;
    const int m0 = blockIdx.y * TILE;
    const int n0 = blockIdx.z * TILE;
    const int tid  = threadIdx.x;
    const int lane = tid & 31;
    const int wave = tid >> 5;
    const int mw = (wave >> 1) * 32;
    const int nw = (wave & 1) * 32;

    v8f acc[2][2];
#pragma unroll
    for (int i = 0; i < 2; ++i)
#pragma unroll
        for (int j = 0; j < 2; ++j)
#pragma unroll
            for (int r = 0; r < 8; ++r) acc[i][j][r] = 0.0f;

    for (int i = tid; i < LC; i += 128) cbuf[i] = 0.0f;

    const size_t chunk_base = (size_t)chunk * LC;

#if TDM_OK
    const unsigned ldsA[2] = { (unsigned)(unsigned long long)(void*)&Alds[0][0],
                               (unsigned)(unsigned long long)(void*)&Alds[1][0] };
    const unsigned ldsB[2] = { (unsigned)(unsigned long long)(void*)&Blds[0][0],
                               (unsigned)(unsigned long long)(void*)&Blds[1][0] };
    const unsigned long long qbase = (unsigned long long)qh;
    const unsigned long long kbase = (unsigned long long)kh;

    auto issue_pair = [&](int buf, int kk) {
        const int b    = kk >> 10;
        const int chi0 = kk & 1023;
        unsigned long long gq = qbase +
            2ull * (((unsigned long long)b * L_ + chunk_base + m0) * HE + chi0);
        unsigned long long gk = kbase +
            2ull * (((unsigned long long)b * L_ + chunk_base + n0) * HE + chi0);
        tdm_load_tile_f16(gq, ldsA[buf]);
        tdm_load_tile_f16(gk, ldsB[buf]);
    };

    if (tid < 32) issue_pair(0, 0);          // prologue: fill buffer 0
#endif

    for (int step = 0; step < NSTEP; ++step) {
        const int cur = step & 1;
        const int kk  = step * KC;
        __syncthreads();   // previous step's LDS consumers done
#if TDM_OK
        if (tid < 32) {    // one wave drives the DMA engine
            if (step + 1 < NSTEP) {
                issue_pair(cur ^ 1, kk + KC);          // prefetch next K-slab
                __builtin_amdgcn_s_wait_tensorcnt(2);  // current pair landed
            } else {
                __builtin_amdgcn_s_wait_tensorcnt(0);  // drain on last step
            }
        }
#else
        // Fallback staging: copy 64x64 f16 tiles with b128 traffic.
        {
            const int b    = kk >> 10;
            const int chi0 = kk & 1023;
#pragma unroll
            for (int i = 0; i < 4; ++i) {
                int f   = tid + i * 128;      // h8 id in tile: 0..511
                int row = f >> 3;             // 0..63
                int c8  = (f & 7) << 3;       // 0..56
                int chi = chi0 + c8;
                *(h8*)&Alds[cur][row * LDST + c8] =
                    *(const h8*)(qh + ((size_t)b * L_ + chunk_base + m0 + row) * HE + chi);
                *(h8*)&Blds[cur][row * LDST + c8] =
                    *(const h8*)(kh + ((size_t)b * L_ + chunk_base + n0 + row) * HE + chi);
            }
        }
#endif
        __syncthreads();   // publish the current buffer

        const _Float16* __restrict__ Ab = &Alds[cur][0];
        const _Float16* __restrict__ Bb = &Blds[cur][0];
#pragma unroll
        for (int ks = 0; ks < KC; ks += 32) {
            const int ka_off = ks + ((lane >> 4) << 3);
            v16h afrag[2], bfrag[2];
#pragma unroll
            for (int t = 0; t < 2; ++t) {
                const _Float16* pa = &Ab[(mw + t * 16 + (lane & 15)) * LDST + ka_off];
                v8h lo = *(const v8h*)pa;
                v8h hi = *(const v8h*)(pa + 16);
                afrag[t] = __builtin_shufflevector(lo, hi,
                    0,1,2,3,4,5,6,7,8,9,10,11,12,13,14,15);
            }
            const int kb_off = ks + ((lane >> 4) << 3);
#pragma unroll
            for (int t = 0; t < 2; ++t) {
                const _Float16* pb = &Bb[(nw + t * 16 + (lane & 15)) * LDST + kb_off];
                v8h lo = *(const v8h*)pb;
                v8h hi = *(const v8h*)(pb + 16);
                bfrag[t] = __builtin_shufflevector(lo, hi,
                    0,1,2,3,4,5,6,7,8,9,10,11,12,13,14,15);
            }
#pragma unroll
            for (int i = 0; i < 2; ++i)
#pragma unroll
                for (int j = 0; j < 2; ++j)
                    acc[i][j] = __builtin_amdgcn_wmma_f32_16x16x32_f16(
                        false, afrag[i], false, bfrag[j],
                        (short)0, acc[i][j], false, false);
        }
    }

    __syncthreads();
    const int mlane = ((lane >> 4) << 3);
#pragma unroll
    for (int i = 0; i < 2; ++i)
#pragma unroll
        for (int j = 0; j < 2; ++j) {
            const int n = n0 + nw + j * 16 + (lane & 15);
            const int mbase = m0 + mw + i * 16 + mlane;
#pragma unroll
            for (int r = 0; r < 8; ++r) {
                int tau = ((mbase + r) - n) & (LC - 1);
                atomicAdd(&cbuf[tau], acc[i][j][r]);   // ds_add_f32
            }
        }
    __syncthreads();
    for (int i = tid; i < LC; i += 128)
        atomicAdd(&corr[chunk * LC + i], cbuf[i]);
}

// ---------------------------------------------------------------------------
// Fallback WMMA GEMM straight from f32 inputs (used if ws too small for the
// f16 copies). Identical math; stages with in-register f32->f16 conversion.
// ---------------------------------------------------------------------------
__global__ __launch_bounds__(128) void corr_gemm_f32_kernel(
        const float* __restrict__ q, const float* __restrict__ k,
        float* __restrict__ corr) {
    __shared__ __attribute__((aligned(16))) _Float16 Alds[TILE * LDST];
    __shared__ __attribute__((aligned(16))) _Float16 Blds[TILE * LDST];
    __shared__ float cbuf[LC];

    const int chunk = blockIdx.x;
    const int m0 = blockIdx.y * TILE;
    const int n0 = blockIdx.z * TILE;
    const int tid  = threadIdx.x;
    const int lane = tid & 31;
    const int wave = tid >> 5;
    const int mw = (wave >> 1) * 32;
    const int nw = (wave & 1) * 32;

    v8f acc[2][2];
#pragma unroll
    for (int i = 0; i < 2; ++i)
#pragma unroll
        for (int j = 0; j < 2; ++j)
#pragma unroll
            for (int r = 0; r < 8; ++r) acc[i][j][r] = 0.0f;

    for (int i = tid; i < LC; i += 128) cbuf[i] = 0.0f;

    const size_t chunk_base = (size_t)chunk * LC;

    for (int kk = 0; kk < CD; kk += KC) {
        __syncthreads();
#pragma unroll
        for (int i = 0; i < 8; ++i) {
            int f   = tid + i * 128;
            int row = f >> 4;
            int c4  = (f & 15) << 2;
            int c   = kk + c4;
            int b   = c >> 10;
            int chi = c & 1023;
            size_t gq = ((size_t)b * L_ + chunk_base + (m0 + row)) * HE + chi;
            size_t gk = ((size_t)b * L_ + chunk_base + (n0 + row)) * HE + chi;
            float4 qa = *(const float4*)(q + gq);
            float4 ka = *(const float4*)(k + gk);
            h4 qv = { (_Float16)qa.x, (_Float16)qa.y, (_Float16)qa.z, (_Float16)qa.w };
            h4 kv = { (_Float16)ka.x, (_Float16)ka.y, (_Float16)ka.z, (_Float16)ka.w };
            *(h4*)&Alds[row * LDST + c4] = qv;
            *(h4*)&Blds[row * LDST + c4] = kv;
        }
        __syncthreads();

#pragma unroll
        for (int ks = 0; ks < KC; ks += 32) {
            const int ka_off = ks + ((lane >> 4) << 3);
            v16h afrag[2], bfrag[2];
#pragma unroll
            for (int t = 0; t < 2; ++t) {
                const _Float16* pa = &Alds[(mw + t * 16 + (lane & 15)) * LDST + ka_off];
                v8h lo = *(const v8h*)pa;
                v8h hi = *(const v8h*)(pa + 16);
                afrag[t] = __builtin_shufflevector(lo, hi,
                    0,1,2,3,4,5,6,7,8,9,10,11,12,13,14,15);
            }
            const int kb_off = ks + ((lane >> 4) << 3);
#pragma unroll
            for (int t = 0; t < 2; ++t) {
                const _Float16* pb = &Blds[(nw + t * 16 + (lane & 15)) * LDST + kb_off];
                v8h lo = *(const v8h*)pb;
                v8h hi = *(const v8h*)(pb + 16);
                bfrag[t] = __builtin_shufflevector(lo, hi,
                    0,1,2,3,4,5,6,7,8,9,10,11,12,13,14,15);
            }
#pragma unroll
            for (int i = 0; i < 2; ++i)
#pragma unroll
                for (int j = 0; j < 2; ++j)
                    acc[i][j] = __builtin_amdgcn_wmma_f32_16x16x32_f16(
                        false, afrag[i], false, bfrag[j],
                        (short)0, acc[i][j], false, false);
        }
    }

    __syncthreads();
    const int mlane = ((lane >> 4) << 3);
#pragma unroll
    for (int i = 0; i < 2; ++i)
#pragma unroll
        for (int j = 0; j < 2; ++j) {
            const int n = n0 + nw + j * 16 + (lane & 15);
            const int mbase = m0 + mw + i * 16 + mlane;
#pragma unroll
            for (int r = 0; r < 8; ++r) {
                int tau = ((mbase + r) - n) & (LC - 1);
                atomicAdd(&cbuf[tau], acc[i][j][r]);
            }
        }
    __syncthreads();
    for (int i = tid; i < LC; i += 128)
        atomicAdd(&corr[chunk * LC + i], cbuf[i]);
}

// ---------------------------------------------------------------------------
__global__ __launch_bounds__(64) void topk_softmax_kernel(
        const float* __restrict__ corr, int* __restrict__ idx,
        float* __restrict__ w) {
    __shared__ float vals[LC];
    const int c = blockIdx.x;
    for (int i = threadIdx.x; i < LC; i += 64)
        vals[i] = corr[c * LC + i] * (1.0f / (float)CD);
    __syncthreads();
    if (threadIdx.x == 0) {
        float tw[TOPK];
        int   ti[TOPK];
        for (int t = 0; t < TOPK; ++t) {
            float best = -3.4e38f;
            int   bi = 0;
            for (int i = 0; i < LC; ++i)
                if (vals[i] > best) { best = vals[i]; bi = i; }
            tw[t] = best; ti[t] = bi;
            vals[bi] = -3.4e38f;
        }
        const float mx = tw[0];
        float e[TOPK], s = 0.0f;
        for (int t = 0; t < TOPK; ++t) { e[t] = __expf(tw[t] - mx); s += e[t]; }
        const float inv = 1.0f / s;
        for (int t = 0; t < TOPK; ++t) {
            w[c * TOPK + t]   = e[t] * inv;
            idx[c * TOPK + t] = ti[t];
        }
    }
}

// ---------------------------------------------------------------------------
__global__ __launch_bounds__(256) void gather_out_kernel(
        const float* __restrict__ v, const int* __restrict__ idx,
        const float* __restrict__ w, float* __restrict__ out) {
    const int bl = blockIdx.x;
    const int b  = bl >> 12;
    const int l  = bl & (L_ - 1);
    const int c  = l >> 9;
    const int t  = l & (LC - 1);
    const int e4 = threadIdx.x;

    float4 acc = make_float4(0.f, 0.f, 0.f, 0.f);
#pragma unroll
    for (int i = 0; i < TOPK; ++i) {
        const int   s  = idx[c * TOPK + i];
        const float wi = w[c * TOPK + i];
        const int   sl = (c << 9) | ((t + s) & (LC - 1));
        const float4 vv =
            *(const float4*)(v + ((size_t)b * L_ + sl) * HE + (size_t)e4 * 4);
        acc.x += wi * vv.x; acc.y += wi * vv.y;
        acc.z += wi * vv.z; acc.w += wi * vv.w;
    }
    *(float4*)(out + ((size_t)b * L_ + l) * HE + (size_t)e4 * 4) = acc;
}

// ---------------------------------------------------------------------------
extern "C" void kernel_launch(void* const* d_in, const int* in_sizes, int n_in,
                              void* d_out, int out_size, void* d_ws, size_t ws_size,
                              hipStream_t stream) {
    (void)in_sizes; (void)n_in; (void)out_size;
    const float* q = (const float*)d_in[0];
    const float* k = (const float*)d_in[1];
    const float* v = (const float*)d_in[2];
    float* out = (float*)d_out;

    const size_t corr_bytes = NC * LC * sizeof(float);
    const size_t need_big = 2 * QH_BYTES + corr_bytes + 2 * NC * TOPK * 8;
    dim3 g(NC, LC / TILE, LC / TILE);

    if (ws_size >= need_big) {
        _Float16* qh = (_Float16*)d_ws;
        _Float16* kh = (_Float16*)((char*)d_ws + QH_BYTES);
        float* corr  = (float*)((char*)d_ws + 2 * QH_BYTES);
        int*   idx   = (int*)((char*)corr + corr_bytes);
        float* w     = (float*)((char*)idx + NC * TOPK * sizeof(int));

        const int n8 = (int)(B_ * (size_t)L_ * HE / 8);
        cvt_f16_kernel<<<(n8 + 255) / 256, 256, 0, stream>>>(q, qh, n8);
        cvt_f16_kernel<<<(n8 + 255) / 256, 256, 0, stream>>>(k, kh, n8);
        zero_corr_kernel<<<(NC * LC + 255) / 256, 256, 0, stream>>>(corr);
        corr_gemm_tdm_kernel<<<g, 128, 0, stream>>>(qh, kh, corr);
        topk_softmax_kernel<<<NC, 64, 0, stream>>>(corr, idx, w);
        gather_out_kernel<<<B_ * L_, 256, 0, stream>>>(v, idx, w, out);
    } else {
        float* corr = (float*)d_ws;
        int*   idx  = (int*)((char*)d_ws + corr_bytes);
        float* w    = (float*)((char*)idx + NC * TOPK * sizeof(int));

        zero_corr_kernel<<<(NC * LC + 255) / 256, 256, 0, stream>>>(corr);
        corr_gemm_f32_kernel<<<g, 128, 0, stream>>>(q, k, corr);
        topk_softmax_kernel<<<NC, 64, 0, stream>>>(corr, idx, w);
        gather_out_kernel<<<B_ * L_, 256, 0, stream>>>(v, idx, w, out);
    }
}
